// CraneForDegree_50998441673258
// MI455X (gfx1250) — compile-verified, matrix operand order checked by reading, not run on hardware
//
#include <hip/hip_runtime.h>

typedef __attribute__((ext_vector_type(16))) _Float16 v16h;
typedef __attribute__((ext_vector_type(8)))  _Float16 v8h;
typedef __attribute__((ext_vector_type(8)))  float    v8f;

namespace {
constexpr int B_   = 512;
constexpr int L_   = 8;
constexpr int DIN_ = 64;
constexpr int H_   = 256;
constexpr int MID_ = 192;
constexpr int E_   = 128;
constexpr float EPS_ = 1e-5f;
}

// ---------------------------------------------------------------------------
// Elementwise prep kernels
// ---------------------------------------------------------------------------
__global__ void k_f32_to_f16(const float* __restrict__ src, _Float16* __restrict__ dst, int n) {
    int i = blockIdx.x * blockDim.x + threadIdx.x;
    if (i < n) dst[i] = (_Float16)src[i];
}

// Fold BatchNorm (eval) + bias into per-channel affine: y = x*A + C
__global__ void k_affine(const float* __restrict__ b, const float* __restrict__ g,
                         const float* __restrict__ be, const float* __restrict__ m,
                         const float* __restrict__ v, float* __restrict__ A,
                         float* __restrict__ C, int n) {
    int i = blockIdx.x * blockDim.x + threadIdx.x;
    if (i < n) {
        float a = g[i] * rsqrtf(v[i] + EPS_);
        A[i] = a;
        C[i] = (b[i] - m[i]) * a + be[i];
    }
}

// ---------------------------------------------------------------------------
// WMMA f16 16x16x32 K-loop, dual N-tile: one 16x32 f32 tile per wave.
// A is [M,K] row-major f16 (lda halfs), B is weights [N,K] row-major f16
// (ldb halfs) == WMMA B-matrix column-major.
// 16-bit A fragment layout (ISA 7.12.2): lane L(0..15): M=L, K = kh..kh+7 and
// kh+16..kh+23 with kh = (lane>=16)*8. B symmetric with lane -> N.
// The A fragment is shared by both WMMAs (operand reuse), and the two WMMAs
// issue back-to-back so they pipeline in the matrix unit.
// ---------------------------------------------------------------------------
struct Acc2 { v8f t0, t1; };

__device__ __forceinline__ Acc2 wmma_kloop2(const _Float16* __restrict__ A, int lda,
                                            const _Float16* __restrict__ Bm, int ldb,
                                            int m0, int n0, int K, int lane) {
    const int r  = lane & 15;
    const int kh = (lane >> 4) << 3;
    const _Float16* pa  = A  + (size_t)(m0 + r) * lda + kh;
    const _Float16* pb0 = Bm + (size_t)(n0 + r) * ldb + kh;
    const _Float16* pb1 = pb0 + (size_t)16 * ldb;
    v8f a0 = {}, a1 = {};
    for (int k0 = 0; k0 < K; k0 += 32) {
        union { v16h v; v8h h[2]; } fa, fb0, fb1;
        fa.h[0]  = *(const v8h*)(pa  + k0);
        fa.h[1]  = *(const v8h*)(pa  + k0 + 16);
        fb0.h[0] = *(const v8h*)(pb0 + k0);
        fb0.h[1] = *(const v8h*)(pb0 + k0 + 16);
        fb1.h[0] = *(const v8h*)(pb1 + k0);
        fb1.h[1] = *(const v8h*)(pb1 + k0 + 16);
        a0 = __builtin_amdgcn_wmma_f32_16x16x32_f16(false, fa.v, false, fb0.v,
                                                    (short)0, a0, false, false);
        a1 = __builtin_amdgcn_wmma_f32_16x16x32_f16(false, fa.v, false, fb1.v,
                                                    (short)0, a1, false, false);
    }
    return {a0, a1};
}

// D layout: element i of v8f -> (M = m0 + (lane>=16)*8 + i, N = n0 + (lane&15))

// GEMM1: h1[l][b][h] = relu(affine( X[b]·W1[l]^T )), K=DIN
__global__ __launch_bounds__(128)
void k_gemm1(const _Float16* __restrict__ Xh,
             const _Float16* __restrict__ W1s, const _Float16* __restrict__ W1d,
             const float* __restrict__ A1s, const float* __restrict__ C1s,
             const float* __restrict__ A1d, const float* __restrict__ C1d,
             _Float16* __restrict__ h1s, _Float16* __restrict__ h1d) {
    const int lane = threadIdx.x & 31, wave = threadIdx.x >> 5;
    const int n0 = blockIdx.x * 64 + (wave & 1) * 32;
    const int m0 = blockIdx.y * 32 + (wave >> 1) * 16;
    const int net = blockIdx.z >> 3, l = blockIdx.z & 7;

    const _Float16* W = (net ? W1d : W1s) + (size_t)l * H_ * DIN_;
    const float* Aaf = (net ? A1d : A1s) + l * H_;
    const float* Caf = (net ? C1d : C1s) + l * H_;
    _Float16* out = (net ? h1d : h1s) + (size_t)l * B_ * H_;

    Acc2 acc = wmma_kloop2(Xh, DIN_, W, DIN_, m0, n0, DIN_, lane);

    const int n = n0 + (lane & 15);
    const float a0 = Aaf[n],      c0 = Caf[n];
    const float a1 = Aaf[n + 16], c1 = Caf[n + 16];
    const int mb = m0 + ((lane >> 4) << 3);
#pragma unroll
    for (int i = 0; i < 8; ++i) {
        float v0 = acc.t0[i] * a0 + c0;
        float v1 = acc.t1[i] * a1 + c1;
        out[(size_t)(mb + i) * H_ + n]      = (_Float16)(v0 > 0.f ? v0 : 0.f);
        out[(size_t)(mb + i) * H_ + n + 16] = (_Float16)(v1 > 0.f ? v1 : 0.f);
    }
}

// GEMM2: h2[l][b][m] = relu(affine( h1[l][b]·W2[l]^T )), K=H
__global__ __launch_bounds__(128)
void k_gemm2(const _Float16* __restrict__ h1s, const _Float16* __restrict__ h1d,
             const _Float16* __restrict__ W2s, const _Float16* __restrict__ W2d,
             const float* __restrict__ A2s, const float* __restrict__ C2s,
             const float* __restrict__ A2d, const float* __restrict__ C2d,
             _Float16* __restrict__ h2s, _Float16* __restrict__ h2d) {
    const int lane = threadIdx.x & 31, wave = threadIdx.x >> 5;
    const int n0 = blockIdx.x * 64 + (wave & 1) * 32;
    const int m0 = blockIdx.y * 32 + (wave >> 1) * 16;
    const int net = blockIdx.z >> 3, l = blockIdx.z & 7;

    const _Float16* A = (net ? h1d : h1s) + (size_t)l * B_ * H_;
    const _Float16* W = (net ? W2d : W2s) + (size_t)l * MID_ * H_;
    const float* Aaf = (net ? A2d : A2s) + l * MID_;
    const float* Caf = (net ? C2d : C2s) + l * MID_;
    _Float16* out = (net ? h2d : h2s) + (size_t)l * B_ * MID_;

    Acc2 acc = wmma_kloop2(A, H_, W, H_, m0, n0, H_, lane);

    const int n = n0 + (lane & 15);
    const float a0 = Aaf[n],      c0 = Caf[n];
    const float a1 = Aaf[n + 16], c1 = Caf[n + 16];
    const int mb = m0 + ((lane >> 4) << 3);
#pragma unroll
    for (int i = 0; i < 8; ++i) {
        float v0 = acc.t0[i] * a0 + c0;
        float v1 = acc.t1[i] * a1 + c1;
        out[(size_t)(mb + i) * MID_ + n]      = (_Float16)(v0 > 0.f ? v0 : 0.f);
        out[(size_t)(mb + i) * MID_ + n + 16] = (_Float16)(v1 > 0.f ? v1 : 0.f);
    }
}

// GEMM3: invE[l][b][e] = 1 / softplus( h2[l][b]·W3[l]^T + b3 ), K=MID
__global__ __launch_bounds__(128)
void k_gemm3(const _Float16* __restrict__ h2s, const _Float16* __restrict__ h2d,
             const _Float16* __restrict__ W3s, const _Float16* __restrict__ W3d,
             const float* __restrict__ b3s, const float* __restrict__ b3d,
             float* __restrict__ invEs, float* __restrict__ invEd) {
    const int lane = threadIdx.x & 31, wave = threadIdx.x >> 5;
    const int n0 = blockIdx.x * 64 + (wave & 1) * 32;
    const int m0 = blockIdx.y * 32 + (wave >> 1) * 16;
    const int net = blockIdx.z >> 3, l = blockIdx.z & 7;

    const _Float16* A = (net ? h2d : h2s) + (size_t)l * B_ * MID_;
    const _Float16* W = (net ? W3d : W3s) + (size_t)l * E_ * MID_;
    const float* b3 = (net ? b3d : b3s) + l * E_;
    float* out = (net ? invEd : invEs) + (size_t)l * B_ * E_;

    Acc2 acc = wmma_kloop2(A, MID_, W, MID_, m0, n0, MID_, lane);

    const int n = n0 + (lane & 15);
    const float bias0 = b3[n], bias1 = b3[n + 16];
    const int mb = m0 + ((lane >> 4) << 3);
#pragma unroll
    for (int i = 0; i < 8; ++i) {
        float x0 = acc.t0[i] + bias0;
        float x1 = acc.t1[i] + bias1;
        float sp0 = (x0 > 20.f) ? x0 : log1pf(__expf(x0));   // softplus, stable
        float sp1 = (x1 > 20.f) ? x1 : log1pf(__expf(x1));
        out[(size_t)(mb + i) * E_ + n]      = 1.0f / sp0;    // store reciprocal
        out[(size_t)(mb + i) * E_ + n + 16] = 1.0f / sp1;
    }
}

// ---------------------------------------------------------------------------
// Min over (r,c) of mem[l,r,c] * invS[b,l,r] * invD[b,l,c].
// mem[l] (64KB) staged in LDS per block; 8 waves = 8 batch rows per block.
// invS is preloaded once per lane (float4) and broadcast with __shfl so the
// hot loop has zero global loads: 1 LDS float4 read + VALU mul/min per row.
// ---------------------------------------------------------------------------
__global__ __launch_bounds__(256)
void k_minratio(const float* __restrict__ mem, const float* __restrict__ invS,
                const float* __restrict__ invD, float* __restrict__ rmin) {
    __shared__ float4 smem4[E_ * E_ / 4];           // 64 KB
    float* smem = (float*)smem4;
    const int l  = blockIdx.x >> 6;                 // 0..7
    const int bg = blockIdx.x & 63;                 // 0..63

    const float4* msrc = (const float4*)(mem + (size_t)l * E_ * E_);
    for (int i = threadIdx.x; i < E_ * E_ / 4; i += blockDim.x) smem4[i] = msrc[i];
    __syncthreads();

    const int wave = threadIdx.x >> 5, lane = threadIdx.x & 31;
    const int b = bg * 8 + wave;
    const float* is = invS + ((size_t)l * B_ + b) * E_;
    const float* id = invD + ((size_t)l * B_ + b) * E_;
    const int c0 = lane * 4;
    const float4 d4 = *(const float4*)(id + c0);    // 4 columns per lane
    const float4 sv = *(const float4*)(is + c0);    // lane q holds invS[4q..4q+3]

    float acc = 3.4e38f;
    for (int q = 0; q < 32; ++q) {                  // rows 4q..4q+3
        float s0 = __shfl(sv.x, q, 32);
        float s1 = __shfl(sv.y, q, 32);
        float s2 = __shfl(sv.z, q, 32);
        float s3 = __shfl(sv.w, q, 32);
        const float* row = smem + (q * 4) * E_ + c0;
        {
            const float4 mv = *(const float4*)(row);
            float t = fminf(fminf(mv.x * d4.x, mv.y * d4.y),
                            fminf(mv.z * d4.z, mv.w * d4.w));
            acc = fminf(acc, t * s0);               // invS > 0: scaling commutes with min
        }
        {
            const float4 mv = *(const float4*)(row + E_);
            float t = fminf(fminf(mv.x * d4.x, mv.y * d4.y),
                            fminf(mv.z * d4.z, mv.w * d4.w));
            acc = fminf(acc, t * s1);
        }
        {
            const float4 mv = *(const float4*)(row + 2 * E_);
            float t = fminf(fminf(mv.x * d4.x, mv.y * d4.y),
                            fminf(mv.z * d4.z, mv.w * d4.w));
            acc = fminf(acc, t * s2);
        }
        {
            const float4 mv = *(const float4*)(row + 3 * E_);
            float t = fminf(fminf(mv.x * d4.x, mv.y * d4.y),
                            fminf(mv.z * d4.z, mv.w * d4.w));
            acc = fminf(acc, t * s3);
        }
    }
#pragma unroll
    for (int off = 16; off > 0; off >>= 1)
        acc = fminf(acc, __shfl_xor(acc, off, 32));
    if (lane == 0) rmin[(size_t)b * L_ + l] = acc;
}

// Decoder: out[b] = decb + sum_{l<=ad} rmin[b,l] * decW[l]
__global__ void k_decode(const float* __restrict__ rmin, const float* __restrict__ decW,
                         const float* __restrict__ decb, const int* __restrict__ ad,
                         float* __restrict__ out) {
    int b = blockIdx.x * blockDim.x + threadIdx.x;
    if (b >= B_) return;
    const int a = ad[0];
    float s = decb[0];
#pragma unroll
    for (int l = 0; l < L_; ++l)
        if (l <= a) s += rmin[b * L_ + l] * decW[l];
    out[b] = s;
}

// ---------------------------------------------------------------------------
extern "C" void kernel_launch(void* const* d_in, const int* in_sizes, int n_in,
                              void* d_out, int out_size, void* d_ws, size_t ws_size,
                              hipStream_t stream) {
    (void)in_sizes; (void)n_in; (void)out_size; (void)ws_size;
    const float* node = (const float*)d_in[0];
    const float* sW1 = (const float*)d_in[1];
    const float* sb1 = (const float*)d_in[2];
    const float* sg1 = (const float*)d_in[3];
    const float* sbe1 = (const float*)d_in[4];
    const float* sm1 = (const float*)d_in[5];
    const float* sv1 = (const float*)d_in[6];
    const float* sW2 = (const float*)d_in[7];
    const float* sb2 = (const float*)d_in[8];
    const float* sg2 = (const float*)d_in[9];
    const float* sbe2 = (const float*)d_in[10];
    const float* sm2 = (const float*)d_in[11];
    const float* sv2 = (const float*)d_in[12];
    const float* sW3 = (const float*)d_in[13];
    const float* sb3 = (const float*)d_in[14];
    const float* dW1 = (const float*)d_in[15];
    const float* db1 = (const float*)d_in[16];
    const float* dg1 = (const float*)d_in[17];
    const float* dbe1 = (const float*)d_in[18];
    const float* dm1 = (const float*)d_in[19];
    const float* dv1 = (const float*)d_in[20];
    const float* dW2 = (const float*)d_in[21];
    const float* db2 = (const float*)d_in[22];
    const float* dg2 = (const float*)d_in[23];
    const float* dbe2 = (const float*)d_in[24];
    const float* dm2 = (const float*)d_in[25];
    const float* dv2 = (const float*)d_in[26];
    const float* dW3 = (const float*)d_in[27];
    const float* db3 = (const float*)d_in[28];
    const float* memM = (const float*)d_in[29];
    const float* decW = (const float*)d_in[30];
    const float* decb = (const float*)d_in[31];
    const int*   adim = (const int*)d_in[32];

    // workspace carve-up (256B aligned slabs)
    char* ws = (char*)d_ws;
    size_t off = 0;
    auto alloc = [&](size_t bytes) -> char* {
        off = (off + 255) & ~(size_t)255;
        char* p = ws + off;
        off += bytes;
        return p;
    };
    _Float16* Xh    = (_Float16*)alloc((size_t)B_ * DIN_ * 2);
    _Float16* W1h[2] = { (_Float16*)alloc((size_t)L_ * H_ * DIN_ * 2),
                         (_Float16*)alloc((size_t)L_ * H_ * DIN_ * 2) };
    _Float16* W2h[2] = { (_Float16*)alloc((size_t)L_ * MID_ * H_ * 2),
                         (_Float16*)alloc((size_t)L_ * MID_ * H_ * 2) };
    _Float16* W3h[2] = { (_Float16*)alloc((size_t)L_ * E_ * MID_ * 2),
                         (_Float16*)alloc((size_t)L_ * E_ * MID_ * 2) };
    float* A1[2] = { (float*)alloc((size_t)L_ * H_ * 4), (float*)alloc((size_t)L_ * H_ * 4) };
    float* C1[2] = { (float*)alloc((size_t)L_ * H_ * 4), (float*)alloc((size_t)L_ * H_ * 4) };
    float* A2[2] = { (float*)alloc((size_t)L_ * MID_ * 4), (float*)alloc((size_t)L_ * MID_ * 4) };
    float* C2[2] = { (float*)alloc((size_t)L_ * MID_ * 4), (float*)alloc((size_t)L_ * MID_ * 4) };
    _Float16* h1[2] = { (_Float16*)alloc((size_t)L_ * B_ * H_ * 2),
                        (_Float16*)alloc((size_t)L_ * B_ * H_ * 2) };
    _Float16* h2[2] = { (_Float16*)alloc((size_t)L_ * B_ * MID_ * 2),
                        (_Float16*)alloc((size_t)L_ * B_ * MID_ * 2) };
    float* invE[2] = { (float*)alloc((size_t)L_ * B_ * E_ * 4),
                       (float*)alloc((size_t)L_ * B_ * E_ * 4) };
    float* rmin = (float*)alloc((size_t)B_ * L_ * 4);

    auto cvt = [&](const float* src, _Float16* dst, int n) {
        k_f32_to_f16<<<(n + 255) / 256, 256, 0, stream>>>(src, dst, n);
    };
    cvt(node, Xh, B_ * DIN_);
    cvt(sW1, W1h[0], L_ * H_ * DIN_);   cvt(dW1, W1h[1], L_ * H_ * DIN_);
    cvt(sW2, W2h[0], L_ * MID_ * H_);   cvt(dW2, W2h[1], L_ * MID_ * H_);
    cvt(sW3, W3h[0], L_ * E_ * MID_);   cvt(dW3, W3h[1], L_ * E_ * MID_);

    k_affine<<<(L_ * H_ + 255) / 256, 256, 0, stream>>>(sb1, sg1, sbe1, sm1, sv1, A1[0], C1[0], L_ * H_);
    k_affine<<<(L_ * H_ + 255) / 256, 256, 0, stream>>>(db1, dg1, dbe1, dm1, dv1, A1[1], C1[1], L_ * H_);
    k_affine<<<(L_ * MID_ + 255) / 256, 256, 0, stream>>>(sb2, sg2, sbe2, sm2, sv2, A2[0], C2[0], L_ * MID_);
    k_affine<<<(L_ * MID_ + 255) / 256, 256, 0, stream>>>(db2, dg2, dbe2, dm2, dv2, A2[1], C2[1], L_ * MID_);

    k_gemm1<<<dim3(H_ / 64, B_ / 32, 2 * L_), 128, 0, stream>>>(
        Xh, W1h[0], W1h[1], A1[0], C1[0], A1[1], C1[1], h1[0], h1[1]);
    k_gemm2<<<dim3(MID_ / 64, B_ / 32, 2 * L_), 128, 0, stream>>>(
        h1[0], h1[1], W2h[0], W2h[1], A2[0], C2[0], A2[1], C2[1], h2[0], h2[1]);
    k_gemm3<<<dim3(E_ / 64, B_ / 32, 2 * L_), 128, 0, stream>>>(
        h2[0], h2[1], W3h[0], W3h[1], sb3, db3, invE[0], invE[1]);

    k_minratio<<<dim3(L_ * (B_ / 8)), 256, 0, stream>>>(memM, invE[0], invE[1], rmin);
    k_decode<<<dim3(2), 256, 0, stream>>>(rmin, decW, decb, adim, (float*)d_out);
}
